// MessageUpdateFunction_24644522344813
// MI455X (gfx1250) — matrix operand order repaired and checked by reading, not compiled, not run on hardware
//
#include <hip/hip_runtime.h>
#include <hip/hip_bf16.h>

// Edge-MLP: out = relu(cat(v_i,v_j,e_ij) @ W1 + b1) @ W2 + b2
// Memory-bound (~32 FLOP/B, ~896 MB total => ~38us floor at 23.3 TB/s).
// Strategy: stream fp32 features once (non-temporal), bf16 weights resident
// in LDS, all math via v_wmma_f32_16x16x32_bf16 (fp32 accumulate).
// Layer1->Layer2 hidden staging is wave-private: sync with s_wait_dscnt
// instead of a block-wide barrier.

typedef __attribute__((ext_vector_type(16))) __bf16 v16bf;
typedef __attribute__((ext_vector_type(8)))  __bf16 v8bf;
typedef __attribute__((ext_vector_type(8)))  float  v8f;
typedef __attribute__((ext_vector_type(4)))  float  v4f;

#define IN_C  64
#define IN_E  32
#define D_IN  160   // 2*IN_C + IN_E
#define HID   64
#define OUT_C 64

// LDS row padding: keep 16B (b128) alignment while perturbing bank mapping
#define W1_PAD (D_IN + 16)   // 176 elems -> 352B rows
#define W2_PAD (HID  + 16)   // 80  elems -> 160B rows
#define H_PAD  (HID  + 8)    // 72  elems -> 144B rows

__global__ __launch_bounds__(256, 2)
void edge_mlp_wmma(const float* __restrict__ v_i,
                   const float* __restrict__ v_j,
                   const float* __restrict__ e_ij,
                   const float* __restrict__ W1,
                   const float* __restrict__ b1,
                   const float* __restrict__ W2,
                   const float* __restrict__ b2,
                   float* __restrict__ out,
                   int E)
{
    __shared__ __bf16 sW1[HID][W1_PAD];    // transposed: [n][k], k = 0..159
    __shared__ __bf16 sW2[OUT_C][W2_PAD];  // transposed: [n][k], k = 0..63
    __shared__ __bf16 sH[8][16][H_PAD];    // per-wave hidden tile, row-major [m][k]

    const int tid = threadIdx.x;

    // Cooperative load + transpose + fp32->bf16 of the (tiny) weights.
    for (int i = tid; i < D_IN * HID; i += 256) {
        int k = i / HID, n = i - k * HID;      // W1 is [k][n] row-major
        sW1[n][k] = (__bf16)W1[i];
    }
    for (int i = tid; i < HID * OUT_C; i += 256) {
        int k = i / OUT_C, n = i - k * OUT_C;  // W2 is [k][n] row-major
        sW2[n][k] = (__bf16)W2[i];
    }
    __syncthreads();   // cross-wave: weights must be visible to every wave

    const int lane = tid & 31;
    const int wave = tid >> 5;
    const int col  = lane & 15;   // N index (and M row for A-side loads)
    const int hi   = lane >> 4;   // half-wave selector
    const int klo  = hi ? 16 : 0; // A-layout: contiguous K run offset for this half

    const long m0   = (long)blockIdx.x * 128 + (long)wave * 16;
    const long row  = m0 + col;                       // edge this lane loads
    const long lrow = (row < (long)E) ? row : (long)(E - 1);  // clamp: keep EXEC all-1s

    // ---------------- Layer 1: [16 x 160] @ [160 x 64] + b1 ----------------
    v8f acc[4];
    #pragma unroll
    for (int nc = 0; nc < 4; ++nc) {
        float bv = b1[nc * 16 + col];          // bias folded into C init
        #pragma unroll
        for (int r = 0; r < 8; ++r) acc[nc][r] = bv;
    }

    // K chunks of 32: kc 0,1 -> v_i ; 2,3 -> v_j ; 4 -> e_ij (boundaries align)
    const float* const base_i = v_i  + lrow * IN_C;
    const float* const base_j = v_j  + lrow * IN_C;
    const float* const base_e = e_ij + lrow * IN_E;
    const float* srcs[5] = { base_i, base_i + 32, base_j, base_j + 32, base_e };

    #pragma unroll
    for (int kc = 0; kc < 5; ++kc) {
        // A fragment (contiguous K layout, matches B): K=[klo..klo+15]
        // Single-touch streaming data: non-temporal loads (TH_NT) keep L2 for weights.
        const float* p = srcs[kc] + klo;
        v4f f0 = __builtin_nontemporal_load((const v4f*)(p));
        v4f f1 = __builtin_nontemporal_load((const v4f*)(p + 4));
        v4f f2 = __builtin_nontemporal_load((const v4f*)(p + 8));
        v4f f3 = __builtin_nontemporal_load((const v4f*)(p + 12));
        v16bf a;
        #pragma unroll
        for (int e = 0; e < 4; ++e) {
            a[e]      = (__bf16)f0[e];
            a[e + 4]  = (__bf16)f1[e];
            a[e + 8]  = (__bf16)f2[e];
            a[e + 12] = (__bf16)f3[e];
        }
        const int kb = kc * 32 + hi * 16;      // B layout: lane>=16 holds K=16..31
        #pragma unroll
        for (int nc = 0; nc < 4; ++nc) {
            v16bf b = *(const v16bf*)&sW1[nc * 16 + col][kb];
            acc[nc] = __builtin_amdgcn_wmma_f32_16x16x32_bf16(
                false, a, false, b, (short)0, acc[nc], false, false);
        }
    }

    // ReLU, then stage hidden tile into LDS in row-major [m][k] (A-friendly)
    #pragma unroll
    for (int nc = 0; nc < 4; ++nc) {
        #pragma unroll
        for (int r = 0; r < 8; ++r) {
            float v = acc[nc][r];
            sH[wave][hi * 8 + r][nc * 16 + col] = (__bf16)(v > 0.0f ? v : 0.0f);
        }
    }
    // sH tile is WAVE-private: only need this wave's DS ops complete + a
    // compiler ordering fence. Avoids a block-wide s_barrier.
    __builtin_amdgcn_wave_barrier();
    asm volatile("s_wait_dscnt 0" ::: "memory");
    __builtin_amdgcn_wave_barrier();

    // ---------------- Layer 2: [16 x 64] @ [64 x 64] + b2 ----------------
    v8f acc2[4];
    #pragma unroll
    for (int nc = 0; nc < 4; ++nc) {
        float bv = b2[nc * 16 + col];
        #pragma unroll
        for (int r = 0; r < 8; ++r) acc2[nc][r] = bv;
    }

    #pragma unroll
    for (int kc = 0; kc < 2; ++kc) {
        const __bf16* hp = &sH[wave][col][kc * 32 + klo];
        v8bf h0 = *(const v8bf*)(hp);
        v8bf h1 = *(const v8bf*)(hp + 8);
        v16bf a;
        #pragma unroll
        for (int e = 0; e < 8; ++e) { a[e] = h0[e]; a[e + 8] = h1[e]; }
        const int kb = kc * 32 + hi * 16;
        #pragma unroll
        for (int nc = 0; nc < 4; ++nc) {
            v16bf b = *(const v16bf*)&sW2[nc * 16 + col][kb];
            acc2[nc] = __builtin_amdgcn_wmma_f32_16x16x32_bf16(
                false, a, false, b, (short)0, acc2[nc], false, false);
        }
    }

    // Store fp32 output (C layout: VGPR r -> M=hi*8+r, lane -> N).
    // Output is written once and not re-read: non-temporal stores.
    #pragma unroll
    for (int r = 0; r < 8; ++r) {
        long m = m0 + hi * 8 + r;
        if (m < (long)E) {
            float* o = out + m * OUT_C + col;
            #pragma unroll
            for (int nc = 0; nc < 4; ++nc)
                __builtin_nontemporal_store(acc2[nc][r], o + nc * 16);
        }
    }
}

extern "C" void kernel_launch(void* const* d_in, const int* in_sizes, int n_in,
                              void* d_out, int out_size, void* d_ws, size_t ws_size,
                              hipStream_t stream) {
    const float* v_i  = (const float*)d_in[0];
    const float* v_j  = (const float*)d_in[1];
    const float* e_ij = (const float*)d_in[2];
    const float* W1   = (const float*)d_in[3];
    const float* b1   = (const float*)d_in[4];
    const float* W2   = (const float*)d_in[5];
    const float* b2   = (const float*)d_in[6];
    float* out = (float*)d_out;

    const int E = in_sizes[0] / IN_C;             // v_i is [E, 64]
    const int blocks = (E + 127) / 128;           // 128 edges per block (8 waves x 16)
    edge_mlp_wmma<<<blocks, 256, 0, stream>>>(v_i, v_j, e_ij, W1, b1, W2, b2, out, E);
}
